// PreTrainedProbabilisticVQ_62569083568266
// MI455X (gfx1250) — matrix operand order repaired and checked by reading, not compiled, not run on hardware
//
#include <hip/hip_runtime.h>

#define DEPTH  4
#define NMIX   8192
#define CH     512
#define MTOT   16384          // 4 * 4096 tokens
#define KSTEPS (CH / 32)      // 16 WMMA K-steps of 32

typedef __attribute__((ext_vector_type(16))) __bf16 v16bf;
typedef __attribute__((ext_vector_type(8)))  float  v8f;

union Frag {
  v16bf v;
  unsigned short us[16];
  uint4 q[2];
};

__device__ __forceinline__ unsigned short f2bf_rn(float x) {
  unsigned int u = __float_as_uint(x);
  u += 0x7FFFu + ((u >> 16) & 1u);     // round-to-nearest-even to bf16
  return (unsigned short)(u >> 16);
}
__device__ __forceinline__ float bf2f(unsigned short h) {
  return __uint_as_float(((unsigned int)h) << 16);
}

// ---------------------------------------------------------------------------
// One wave per codebook row: emit bf16 hi/lo planes and bias = -0.5*|m|^2.
// ---------------------------------------------------------------------------
__global__ __launch_bounds__(32) void prep_codebook(
    const float* __restrict__ mus, unsigned short* __restrict__ mh,
    unsigned short* __restrict__ ml, float* __restrict__ bias) {
  const int row  = blockIdx.x;          // d*NMIX + v
  const int lane = threadIdx.x;         // 0..31, 16 channels each
  const float* src = mus + (size_t)row * CH;
  float acc = 0.f;
#pragma unroll
  for (int g = 0; g < 2; ++g) {
    const int c0 = lane * 16 + g * 8;
    float4 a = *(const float4*)(src + c0);
    float4 b = *(const float4*)(src + c0 + 4);
    float xs[8] = {a.x, a.y, a.z, a.w, b.x, b.y, b.z, b.w};
    union { unsigned short us[8]; uint4 q; } hq, lq;
#pragma unroll
    for (int e = 0; e < 8; ++e) {
      float f = xs[e];
      unsigned short h = f2bf_rn(f);
      hq.us[e] = h;
      lq.us[e] = f2bf_rn(f - bf2f(h));
      acc += f * f;
    }
    *(uint4*)(mh + (size_t)row * CH + c0) = hq.q;
    *(uint4*)(ml + (size_t)row * CH + c0) = lq.q;
  }
#pragma unroll
  for (int m = 16; m >= 1; m >>= 1) acc += __shfl_xor(acc, m, 32);
  if (lane == 0) bias[row] = -0.5f * acc;
}

// ---------------------------------------------------------------------------
// Fused bf16x3 WMMA GEMM + row argmax.
// Block = 256 threads = 8 waves; wave w owns token rows [mb, mb+16).
// A (residual) fragments for full K=512 live in registers (hi+lo).
// Sweep all V codewords in 16-column tiles; accumulator seeded with -0.5|m|^2
// so score = r.m - 0.5|m|^2 and argmax == reference argmin of distance.
// ---------------------------------------------------------------------------
__global__ __launch_bounds__(256, 1) void rvq_argmax(
    const float* __restrict__ r, const unsigned short* __restrict__ mh,
    const unsigned short* __restrict__ ml, const float* __restrict__ bias,
    int* __restrict__ ids) {
  const int wave = threadIdx.x >> 5;
  const int lane = threadIdx.x & 31;
  const int lrow = lane & 15;
  const int half = lane >> 4;           // 0: K group low, 1: K group high
  const int s    = half * 8;            // A-fragment 8+8 K split per ISA layout
  const int mb   = blockIdx.x * 128 + wave * 16;

  // ---- Fill A fragments (hi/lo bf16) for all 16 K-steps into registers ----
  Frag Ah[KSTEPS], Al[KSTEPS];
  const float* rbase = r + (size_t)(mb + lrow) * CH;
#pragma unroll
  for (int ks = 0; ks < KSTEPS; ++ks) {
    const int kb = ks * 32;
#pragma unroll
    for (int g = 0; g < 2; ++g) {       // elements g*8..g*8+7 <- K = kb+g*16+s
      float4 x0 = *(const float4*)(rbase + kb + g * 16 + s);
      float4 x1 = *(const float4*)(rbase + kb + g * 16 + s + 4);
      float xs[8] = {x0.x, x0.y, x0.z, x0.w, x1.x, x1.y, x1.z, x1.w};
#pragma unroll
      for (int e = 0; e < 8; ++e) {
        unsigned short h = f2bf_rn(xs[e]);
        Ah[ks].us[g * 8 + e] = h;
        Al[ks].us[g * 8 + e] = f2bf_rn(xs[e] - bf2f(h));
      }
    }
  }

  float best[8];
  int   bidx[8];
#pragma unroll
  for (int j = 0; j < 8; ++j) { best[j] = -3.402823466e+38f; bidx[j] = 0; }

  // ---- Sweep codebook ----
  for (int nb = 0; nb < NMIX; nb += 16) {
    const int v = nb + lrow;
    const unsigned short* mhb = mh + (size_t)v * CH;
    const unsigned short* mlb = ml + (size_t)v * CH;
    __builtin_prefetch(mhb + 16 * CH, 0, 0);
    __builtin_prefetch(mlb + 16 * CH, 0, 0);

    const float b = bias[v];
    v8f c;
#pragma unroll
    for (int j = 0; j < 8; ++j) c[j] = b;

#pragma unroll
    for (int ks = 0; ks < KSTEPS; ++ks) {
      const int kb = ks * 32;
      // B 32x16 bf16 layout: same 8+8 K split as A (K = kb+g*16+half*8).
      Frag Bh, Bl;
      Bh.q[0] = *(const uint4*)(mhb + kb + half * 8);
      Bh.q[1] = *(const uint4*)(mhb + kb + 16 + half * 8);
      Bl.q[0] = *(const uint4*)(mlb + kb + half * 8);
      Bl.q[1] = *(const uint4*)(mlb + kb + 16 + half * 8);
      c = __builtin_amdgcn_wmma_f32_16x16x32_bf16(false, Ah[ks].v, false, Bh.v,
                                                  (short)0, c, false, false);
      c = __builtin_amdgcn_wmma_f32_16x16x32_bf16(false, Ah[ks].v, false, Bl.v,
                                                  (short)0, c, false, false);
      c = __builtin_amdgcn_wmma_f32_16x16x32_bf16(false, Al[ks].v, false, Bh.v,
                                                  (short)0, c, false, false);
    }
#pragma unroll
    for (int j = 0; j < 8; ++j) {
      if (c[j] > best[j]) { best[j] = c[j]; bidx[j] = v; }
    }
  }

  // ---- Cross-lane argmax within each 16-lane half (rows fixed per half) ----
#pragma unroll
  for (int j = 0; j < 8; ++j) {
    float bv = best[j];
    int   bi = bidx[j];
#pragma unroll
    for (int m = 8; m >= 1; m >>= 1) {
      float ov = __shfl_xor(bv, m, 32);
      int   oi = __shfl_xor(bi, m, 32);
      if (ov > bv || (ov == bv && oi < bi)) { bv = ov; bi = oi; }
    }
    if (lrow == 0) ids[mb + half * 8 + j] = bi;
  }
}

// ---------------------------------------------------------------------------
// r[t] -= mus_d[ids_d[t]]
// ---------------------------------------------------------------------------
__global__ __launch_bounds__(128) void rvq_update(
    float* __restrict__ r, const float* __restrict__ mus_d,
    const int* __restrict__ ids_d) {
  const int t  = blockIdx.x;
  const int c4 = threadIdx.x;           // 128 threads * float4 = 512 channels
  const int idx = ids_d[t];
  const float4 m = *(const float4*)(mus_d + (size_t)idx * CH + c4 * 4);
  float4* rp = (float4*)(r + (size_t)t * CH) + c4;
  float4 rv = *rp;
  rv.x -= m.x; rv.y -= m.y; rv.z -= m.z; rv.w -= m.w;
  *rp = rv;
}

__global__ void write_ids(const int* __restrict__ ids, float* __restrict__ out) {
  const int i = blockIdx.x * 256 + threadIdx.x;
  if (i < DEPTH * MTOT) out[i] = (float)ids[i];
}

__global__ void write_zq(const float* __restrict__ z, const float* __restrict__ r,
                         float* __restrict__ out) {
  const int i = blockIdx.x * 256 + threadIdx.x;     // float4 index
  const float4 zv = ((const float4*)z)[i];
  const float4 rv = ((const float4*)r)[i];
  float4 o;
  o.x = zv.x - rv.x; o.y = zv.y - rv.y; o.z = zv.z - rv.z; o.w = zv.w - rv.w;
  ((float4*)(out + DEPTH * MTOT))[i] = o;
}

extern "C" void kernel_launch(void* const* d_in, const int* in_sizes, int n_in,
                              void* d_out, int out_size, void* d_ws, size_t ws_size,
                              hipStream_t stream) {
  (void)in_sizes; (void)n_in; (void)out_size; (void)ws_size;
  const float* z   = (const float*)d_in[0];   // [4,4096,512] fp32
  const float* mus = (const float*)d_in[1];   // [4,8192,512] fp32
  float* out = (float*)d_out;

  char* ws = (char*)d_ws;
  float* r = (float*)ws;                    ws += (size_t)MTOT * CH * 4;          // 32 MB
  unsigned short* mh = (unsigned short*)ws; ws += (size_t)DEPTH * NMIX * CH * 2;  // 32 MB
  unsigned short* ml = (unsigned short*)ws; ws += (size_t)DEPTH * NMIX * CH * 2;  // 32 MB
  float* bias = (float*)ws;                 ws += (size_t)DEPTH * NMIX * 4;
  int* ids = (int*)ws;                                                            // 4*16384

  prep_codebook<<<DEPTH * NMIX, 32, 0, stream>>>(mus, mh, ml, bias);
  hipMemcpyAsync(r, z, (size_t)MTOT * CH * 4, hipMemcpyDeviceToDevice, stream);

  for (int d = 0; d < DEPTH; ++d) {
    rvq_argmax<<<MTOT / 128, 256, 0, stream>>>(
        r, mh + (size_t)d * NMIX * CH, ml + (size_t)d * NMIX * CH,
        bias + d * NMIX, ids + d * MTOT);
    rvq_update<<<MTOT, 128, 0, stream>>>(r, mus + (size_t)d * NMIX * CH,
                                         ids + d * MTOT);
  }

  write_ids<<<(DEPTH * MTOT + 255) / 256, 256, 0, stream>>>(ids, out);
  write_zq<<<(MTOT * CH / 4) / 256, 256, 0, stream>>>(z, r, out);
}
